// DSTCarbonFormer_37357625541344
// MI455X (gfx1250) — matrix operand order repaired and checked by reading, not compiled, not run on hardware
//
#include <hip/hip_runtime.h>
#include <math.h>

typedef __attribute__((ext_vector_type(16))) __bf16 v16bf;
typedef __attribute__((ext_vector_type(8)))  float  v8f;

#define DEV __device__ __forceinline__

DEV int imin(int a,int b){ return a<b?a:b; }
DEV int imax(int a,int b){ return a>b?a:b; }
DEV float geluf(float v){ return 0.5f*v*(1.f+erff(v*0.70710678118f)); }
DEV float actf(float v,int a){
  if(a==1) return geluf(v);
  if(a==2) return 1.f/(1.f+__expf(-v));
  if(a==3) return v/(1.f+__expf(-v));
  return v;
}
DEV unsigned int bf16r(float f){
  unsigned int u = __float_as_uint(f);
  return (u + 0x7FFFu + ((u>>16)&1u))>>16;
}
DEV unsigned int pk_bf16(float a,float b){ return (bf16r(b)<<16)|(bf16r(a)&0xFFFFu); }

// ---------------------------------------------------------------- GEMM (WMMA)
struct ConvGeo {
  int M,N,K,Kp;
  int lda,ldc,outCF,act;
  int Ti,Hi,Wi,Ho,Wo, sd,sh,sw, pd,ph,pw;
};

__global__ void k_tap(int4* tap,int K,int Kp,int kvol,int kh,int kw,int strideC){
  int k = blockIdx.x*blockDim.x + threadIdx.x;
  if (k>=Kp) return;
  if (k>=K){ tap[k]=make_int4(0,0,0,0); return; }
  int c=k/kvol, r=k-c*kvol;
  int kd=r/(kh*kw); int r2=r-kd*kh*kw;
  tap[k]=make_int4(c*strideC, kd, r2/kw, r2%kw);
}

__global__ void k_cvtw(const float* src, unsigned short* dst, int N,int K,int Np,int Kp,int mode){
  long i = blockIdx.x*(long)blockDim.x + threadIdx.x;
  long tot = (long)Np*Kp; if (i>=tot) return;
  int n = (int)(i/Kp), k = (int)(i - (long)n*Kp);
  float v=0.f;
  if (n<N && k<K){
    if (mode==0) v = src[(size_t)n*K + k];
    else if (mode==1) v = src[(size_t)k*N + n];
    else { // [N, I*16] with 4x4 spatial flip
      int i0=k>>4, r=k&15;
      int fr=(3-(r>>2))*4 + (3-(r&3));
      v = src[(size_t)n*K + i0*16 + fr];
    }
  }
  dst[i]=(unsigned short)bf16r(v);
}

// branchless A-element gather: always loads from a clamped address, selects 0.
template<int MODE>
DEV float gload(const float* __restrict__ A, const int4* __restrict__ tap,
                const ConvGeo& g, bool mv, int mcl,int t,int y,int x,int k){
  if (MODE==0){
    int kk = imin(k, g.K-1);
    float v = A[(size_t)mcl*g.lda + kk];
    return (mv & (k<g.K)) ? v : 0.f;
  } else if (MODE==1){
    int4 tp = tap[imin(k, g.K-1)];
    int iz=t*g.sd-g.pd+tp.y, iy=y*g.sh-g.ph+tp.z, ix=x*g.sw-g.pw+tp.w;
    bool ok = mv & (k<g.K) &
              ((unsigned)iz<(unsigned)g.Ti) & ((unsigned)iy<(unsigned)g.Hi) &
              ((unsigned)ix<(unsigned)g.Wi);
    int cz=imin(imax(iz,0),g.Ti-1), cy=imin(imax(iy,0),g.Hi-1), cx=imin(imax(ix,0),g.Wi-1);
    float v = A[(size_t)tp.x + ((size_t)cz*g.Hi+cy)*g.Wi + cx];
    return ok ? v : 0.f;
  } else {
    int4 tp = tap[imin(k, g.K-1)];
    int py=y+tp.z-2, px=x+tp.w-2;
    int hy=py>>1, hx=px>>1;
    bool ok = mv & (k<g.K) & (py>=0) & (px>=0) & ((py&1)==0) & ((px&1)==0) &
              (hy<g.Hi) & (hx<g.Wi);
    int cy=imin(imax(hy,0),g.Hi-1), cx=imin(imax(hx,0),g.Wi-1);
    float v = A[(size_t)tp.x + ((size_t)t*g.Hi+cy)*g.Wi + cx];
    return ok ? v : 0.f;
  }
}

// one wave computes a 16(M) x 64(N) strip: 4 WMMA per K-step, A reused 4x.
template<int MODE>
__global__ void k_gemm_t(const float* __restrict__ A, const unsigned short* __restrict__ Wb,
                         const float* __restrict__ bias, float* __restrict__ out,
                         const int4* __restrict__ tap, ConvGeo g){
  int lane=threadIdx.x, half=lane>>4, l15=lane&15;
  int tm=blockIdx.x*16, tn=blockIdx.y*64;
  int mA=tm+l15;
  bool mv = mA<g.M;
  int mcl = imin(mA, g.M-1);
  int t=0,y=0,x=0;
  if (MODE!=0){ int hw=g.Ho*g.Wo; t=mcl/hw; int r=mcl-t*hw; y=r/g.Wo; x=r-y*g.Wo; }
  const unsigned int* Wp=(const unsigned int*)Wb;
  size_t wrow0 = ((size_t)(tn+l15)*g.Kp)>>1;
  size_t wstep = ((size_t)16*g.Kp)>>1;
  v8f acc0={0.f,0.f,0.f,0.f,0.f,0.f,0.f,0.f};
  v8f acc1=acc0, acc2=acc0, acc3=acc0;
  for (int kb=0; kb<g.Kp; kb+=32){
    union { v16bf v; unsigned int u[8]; } a;
    #pragma unroll
    for (int i=0;i<8;++i){
      int k0 = kb + half*16 + i*2;
      float f0 = gload<MODE>(A,tap,g,mv,mcl,t,y,x,k0);
      float f1 = gload<MODE>(A,tap,g,mv,mcl,t,y,x,k0+1);
      a.u[i]=pk_bf16(f0,f1);
    }
    union { v16bf v; unsigned int u[8]; } b0,b1,b2,b3;
    int kofs = (kb + half*16)>>1;
    #pragma unroll
    for (int i=0;i<8;++i){
      b0.u[i]=Wp[wrow0           + kofs + i];
      b1.u[i]=Wp[wrow0 +   wstep + kofs + i];
      b2.u[i]=Wp[wrow0 + 2*wstep + kofs + i];
      b3.u[i]=Wp[wrow0 + 3*wstep + kofs + i];
    }
    acc0=__builtin_amdgcn_wmma_f32_16x16x32_bf16(false,a.v,false,b0.v,(short)0,acc0,false,false);
    acc1=__builtin_amdgcn_wmma_f32_16x16x32_bf16(false,a.v,false,b1.v,(short)0,acc1,false,false);
    acc2=__builtin_amdgcn_wmma_f32_16x16x32_bf16(false,a.v,false,b2.v,(short)0,acc2,false,false);
    acc3=__builtin_amdgcn_wmma_f32_16x16x32_bf16(false,a.v,false,b3.v,(short)0,acc3,false,false);
  }
  #pragma unroll
  for (int j=0;j<4;++j){
    v8f accv = (j==0)?acc0:((j==1)?acc1:((j==2)?acc2:acc3));
    int gnb = tn + j*16 + l15;
    #pragma unroll
    for (int i=0;i<8;++i){
      int gm=tm+i+half*8;
      if (gm<g.M && gnb<g.N){
        float v=accv[i] + (bias? bias[gnb]:0.f);
        v=actf(v,g.act);
        if (g.outCF) out[(size_t)gnb*g.M + gm]=v;
        else         out[(size_t)gm*g.ldc + gnb]=v;
      }
    }
  }
}

// ---------------------------------------------------------------- elementwise
#define EWIDX int idx=blockIdx.x*blockDim.x+threadIdx.x; if(idx>=n) return;
__global__ void k_scale(float* o,const float* i,float s,int n){ EWIDX o[idx]=i[idx]*s; }
__global__ void k_copy (float* o,const float* i,int n){ EWIDX o[idx]=i[idx]; }
__global__ void k_fill (float* o,float v,int n){ EWIDX o[idx]=v; }
__global__ void k_add  (float* d,const float* s,int n){ EWIDX d[idx]+=s[idx]; }
__global__ void k_add2 (float* d,const float* a,const float* b,int n){ EWIDX d[idx]=a[idx]+b[idx]; }
__global__ void k_axpyg(float* d,const float* s,const float* gate,int e,int n){ EWIDX d[idx]+=gate[e]*s[idx]; }
__global__ void k_sft  (float* o,const float* m,const float* s,const float* h,int n){ EWIDX o[idx]=m[idx]*(1.f+s[idx])+h[idx]; }
__global__ void k_blend(float* x,const float* g,const float* a,int n){ EWIDX float gv=g[idx]; x[idx]=gv*x[idx]+(1.f-gv)*a[idx]; }
__global__ void k_addvec_cf(float* x,const float* v,int Msp,int n){ EWIDX x[idx]+=v[idx/Msp]; }
__global__ void k_cf2rm(float* d,const float* s,int C,int L){ int n=C*L; EWIDX int l=idx/C,c=idx-l*C; d[idx]=s[(size_t)c*L+l]; }
__global__ void k_rm2cf(float* d,const float* s,int C,int L){ int n=C*L; EWIDX int l=idx/C,c=idx-l*C; d[(size_t)c*L+l]=s[idx]; }
__global__ void k_flip (float* d,const float* s,int L,int C){ int n=L*C; EWIDX int l=idx/C,c=idx-l*C; d[idx]=s[(size_t)(L-1-l)*C+c]; }
__global__ void k_addrows(float* xf,const float* u,int L,int C,int flip){ int n=L*C; EWIDX int l=idx/C,c=idx-l*C; int sl=flip?(L-1-l):l; xf[idx]+=u[(size_t)sl*C+c]; }
__global__ void k_mulsiluz(float* y,const float* xz,int L){ int n=L*128; EWIDX int l=idx>>7,d=idx&127; float z=xz[(size_t)l*256+128+d]; y[idx]*=z/(1.f+__expf(-z)); }

// ---------------------------------------------------------------- depthwise
__global__ void k_dw(const float* in,const float* w,float* out,int C,int Ti,int Hi,int Wi,
                     int kd,int kh,int kw,int pd,int ph,int pw,int act){
  int thw=Ti*Hi*Wi, n=C*thw; EWIDX
  int c=idx/thw; int r=idx-c*thw;
  int t=r/(Hi*Wi); r-=t*Hi*Wi; int y=r/Wi; int x=r-y*Wi;
  const float* wc=w+(size_t)c*kd*kh*kw;
  const float* ic=in+(size_t)c*thw;
  float s=0.f;
  for(int a=0;a<kd;++a){ int iz=t-pd+a; if((unsigned)iz>=(unsigned)Ti) continue;
    for(int b=0;b<kh;++b){ int iy=y-ph+b; if((unsigned)iy>=(unsigned)Hi) continue;
      for(int cc=0;cc<kw;++cc){ int ix=x-pw+cc; if((unsigned)ix>=(unsigned)Wi) continue;
        s+=ic[((size_t)iz*Hi+iy)*Wi+ix]*wc[((size_t)a*kh+b)*kw+cc]; } } }
  out[idx]=act?geluf(s):s;
}

// ---------------------------------------------------------------- norms
__global__ void k_gn_stats(const float* x,float* stats,int C,int Msp){
  __shared__ float s1[256],s2[256];
  int g=blockIdx.x, Cg=C/8;
  size_t n=(size_t)Cg*Msp;
  const float* base=x+(size_t)g*Cg*Msp;
  float a=0.f,b=0.f;
  for(size_t i=threadIdx.x;i<n;i+=blockDim.x){ float v=base[i]; a+=v; b+=v*v; }
  s1[threadIdx.x]=a; s2[threadIdx.x]=b; __syncthreads();
  for(int s=128;s>0;s>>=1){ if(threadIdx.x<(unsigned)s){ s1[threadIdx.x]+=s1[threadIdx.x+s]; s2[threadIdx.x]+=s2[threadIdx.x+s]; } __syncthreads(); }
  if(threadIdx.x==0){ float mean=s1[0]/(float)n; float var=s2[0]/(float)n-mean*mean;
    stats[g*2]=mean; stats[g*2+1]=rsqrtf(var+1e-5f); }
}
__global__ void k_gn_apply(float* x,const float* stats,const float* gm,const float* bt,int C,int Msp,int act){
  int n=C*Msp; EWIDX
  int c=idx/Msp; int g=c/(C/8);
  float v=(x[idx]-stats[g*2])*stats[g*2+1]*gm[c]+bt[c];
  x[idx]=act?geluf(v):v;
}
__global__ void k_lnorm(const float* x,float* o,const float* g,const float* b,int L){
  __shared__ float s1[64],s2[64];
  int row=blockIdx.x; int tid=threadIdx.x;
  float v=x[(size_t)row*64+tid];
  s1[tid]=v; s2[tid]=v*v; __syncthreads();
  for(int s=32;s>0;s>>=1){ if(tid<s){ s1[tid]+=s1[tid+s]; s2[tid]+=s2[tid+s]; } __syncthreads(); }
  float mean=s1[0]*(1.f/64.f); float var=s2[0]*(1.f/64.f)-mean*mean;
  o[(size_t)row*64+tid]=(v-mean)*rsqrtf(var+1e-5f)*g[tid]+b[tid];
}

// ---------------------------------------------------------------- reductions
__global__ void k_chan_mean(const float* x,float* o,int Msp){
  __shared__ float s[256];
  int c=blockIdx.x; const float* b=x+(size_t)c*Msp;
  float a=0.f; for(int i=threadIdx.x;i<Msp;i+=blockDim.x) a+=b[i];
  s[threadIdx.x]=a; __syncthreads();
  for(int t=128;t>0;t>>=1){ if(threadIdx.x<(unsigned)t) s[threadIdx.x]+=s[threadIdx.x+t]; __syncthreads(); }
  if(threadIdx.x==0) o[c]=s[0]/(float)Msp;
}
__global__ void k_chan_dot(const float* x,const float* w,float* o,int Msp){
  __shared__ float s[256];
  int c=blockIdx.x; const float* b=x+(size_t)c*Msp;
  float a=0.f; for(int i=threadIdx.x;i<Msp;i+=blockDim.x) a+=b[i]*w[i];
  s[threadIdx.x]=a; __syncthreads();
  for(int t=128;t>0;t>>=1){ if(threadIdx.x<(unsigned)t) s[threadIdx.x]+=s[threadIdx.x+t]; __syncthreads(); }
  if(threadIdx.x==0) o[c]=s[0];
}
__global__ void k_softmax1(float* v,int n){
  __shared__ float red[256];
  int tid=threadIdx.x;
  float mx=-1e30f; for(int i=tid;i<n;i+=256) mx=fmaxf(mx,v[i]);
  red[tid]=mx; __syncthreads();
  for(int s=128;s>0;s>>=1){ if(tid<s) red[tid]=fmaxf(red[tid],red[tid+s]); __syncthreads(); }
  mx=red[0]; __syncthreads();
  float sm=0.f; for(int i=tid;i<n;i+=256){ float e=__expf(v[i]-mx); v[i]=e; sm+=e; }
  red[tid]=sm; __syncthreads();
  for(int s=128;s>0;s>>=1){ if(tid<s) red[tid]+=red[tid+s]; __syncthreads(); }
  float inv=1.f/red[0];
  for(int i=tid;i<n;i+=256) v[i]*=inv;
}
__global__ void k_kproj(const float* x,const float* kw,const float* kb,float* o,int C,int Msp){
  int idx=blockIdx.x*blockDim.x+threadIdx.x; if(idx>=Msp) return;
  float s=kb[0];
  for(int c=0;c<C;++c) s+=x[(size_t)c*Msp+idx]*kw[c];
  o[idx]=s;
}
__global__ void k_moe_gate(const float* pool,const float* gw,const float* gb,float* gate){
  __shared__ float lg[4];
  int tid=threadIdx.x;
  if(tid<4){ float s=gb[tid]; for(int c=0;c<96;++c) s+=pool[c]*gw[c*4+tid]; lg[tid]=s; }
  __syncthreads();
  if(tid==0){
    float mx=fmaxf(fmaxf(lg[0],lg[1]),fmaxf(lg[2],lg[3]));
    float e0=__expf(lg[0]-mx),e1=__expf(lg[1]-mx),e2=__expf(lg[2]-mx),e3=__expf(lg[3]-mx);
    float inv=1.f/(e0+e1+e2+e3);
    gate[0]=e0*inv; gate[1]=e1*inv; gate[2]=e2*inv; gate[3]=e3*inv;
  }
}
__global__ void k_gc_mlp(const float* ctx,const float* tw1,const float* tb1,
                         const float* tw2,const float* tb2,float* t2){
  __shared__ float lc[96],l1[24];
  int tid=threadIdx.x;
  if(tid<96) lc[tid]=ctx[tid];
  __syncthreads();
  if(tid<24){ float s=tb1[tid]; for(int c=0;c<96;++c) s+=lc[c]*tw1[tid*96+c]; l1[tid]=geluf(s); }
  __syncthreads();
  if(tid<96){ float s=tb2[tid]; for(int j=0;j<24;++j) s+=l1[j]*tw2[tid*24+j]; t2[tid]=s; }
}

// ---------------------------------------------------------------- mamba
__global__ void k_cconv(const float* xz,const float* cw,const float* cb,float* xi,int L){
  int n=L*128; EWIDX
  int l=idx>>7, d=idx&127;
  float s=cb[d];
  #pragma unroll
  for(int j=0;j<4;++j){ int ls=l-3+j; if(ls>=0) s+=xz[(size_t)ls*256+d]*cw[d*4+j]; }
  xi[idx]=s/(1.f+__expf(-s));
}
__global__ void k_dt(const float* dbl,const float* dtw,const float* dtb,float* dt,int L){
  int n=L*128; EWIDX
  int l=idx>>7, d=idx&127;
  float s=dtb[d];
  #pragma unroll
  for(int r=0;r<4;++r) s+=dbl[(size_t)l*36+r]*dtw[r*128+d];
  dt[idx]=(s>20.f)?s:log1pf(__expf(s));
}
__global__ void k_scan(const float* dt,const float* dbl,const float* xi,
                       const float* A_log,const float* Dp,float* y,int L){
  int lane=threadIdx.x, half=lane>>4, s=lane&15;
  int d=blockIdx.x*2+half;
  float a=-__expf(A_log[d*16+s]);
  float h=0.f, Dd=Dp[d];
  for(int l=0;l<L;++l){
    float dtv=dt[(size_t)l*128+d];
    float Bv=dbl[(size_t)l*36+4+s];
    float Cv=dbl[(size_t)l*36+20+s];
    float xv=xi[(size_t)l*128+d];
    h=h*__expf(dtv*a)+dtv*Bv*xv;
    float c=h*Cv;
    #pragma unroll
    for(int m=1;m<16;m<<=1) c+=__shfl_xor(c,m,32);
    if(s==0) y[(size_t)l*128+d]=c+xv*Dd;
  }
}

// ---------------------------------------------------------------- final
__global__ void k_final(float* out,const float* mainp,const float* r,const float* aux,int M){
  int n=M; EWIDX
  float mn=mainp[idx]*(1.f/11.f);
  float pred=mn+r[idx];
  float prior=0.5f*(aux[idx]+aux[6*M+idx]);
  float mask=(mn>0.f||prior>1e-6f)?1.f:0.f;
  float o=fmaxf(pred*mask*11.f,0.f);
  out[idx]=o; out[M+idx]=o;
}

// ================================================================= host side
struct GW { unsigned short* wb; int4* tap; };
struct MP { const float *A_log,*D,*conv_b,*conv_w,*dt_b,*dt_w,*in_b,*in_w,*out_b,*out_w,*xproj_w; };
struct MLayer { MP bwd, fwd; const float *ln_b,*ln_g; };
struct MWS { float *xf,*hb,*hf,*upd,*xz,*xi,*dbl,*dt,*y; };

static dim3 grd(long n){ return dim3((unsigned)((n+255)/256)); }

static void gemm_core(hipStream_t st,GW gw,const float* A,const float* Wsrc,int wmode,
                      const float* bias,float* out,ConvGeo g,int amode,
                      int kvol,int kh,int kw,int strideC){
  int Np=(g.N+63)&~63;
  long nw=(long)Np*g.Kp;
  k_cvtw<<<grd(nw),256,0,st>>>(Wsrc,gw.wb,g.N,g.K,Np,g.Kp,wmode);
  if(amode) k_tap<<<grd(g.Kp),256,0,st>>>(gw.tap,g.K,g.Kp,kvol,kh,kw,strideC);
  dim3 grid((g.M+15)/16,Np/64);
  if      (amode==0) k_gemm_t<0><<<grid,32,0,st>>>(A,gw.wb,bias,out,gw.tap,g);
  else if (amode==1) k_gemm_t<1><<<grid,32,0,st>>>(A,gw.wb,bias,out,gw.tap,g);
  else               k_gemm_t<2><<<grid,32,0,st>>>(A,gw.wb,bias,out,gw.tap,g);
}
static void gemm_rm(hipStream_t st,GW gw,const float* A,int lda,const float* Wsrc,int wmode,
                    const float* bias,float* out,int ldc,int M,int N,int K,int act){
  ConvGeo g{}; g.M=M; g.N=N; g.K=K; g.Kp=(K+31)&~31; g.lda=lda; g.ldc=ldc; g.outCF=0; g.act=act;
  gemm_core(st,gw,A,Wsrc,wmode,bias,out,g,0,1,1,1,0);
}
static void gemm_conv(hipStream_t st,GW gw,const float* in,const float* Wsrc,int wmode,
                      const float* bias,float* out,int Cin,int Ti,int Hi,int Wi,
                      int kd,int kh,int kw,int sd,int sh,int sw,int pd,int ph,int pw,
                      int Ho,int Wo,int Cout,int act,int amode){
  ConvGeo g{}; g.M=Ti*Ho*Wo; g.N=Cout; g.K=Cin*kd*kh*kw; g.Kp=(g.K+31)&~31;
  g.outCF=1; g.act=act;
  g.Ti=Ti; g.Hi=Hi; g.Wi=Wi; g.Ho=Ho; g.Wo=Wo; g.sd=sd; g.sh=sh; g.sw=sw; g.pd=pd; g.ph=ph; g.pw=pw;
  gemm_core(st,gw,in,Wsrc,wmode,bias,out,g,amode,kd*kh*kw,kh,kw,Ti*Hi*Wi);
}

static void run_mamba(hipStream_t st,GW gw,MWS& w,const float* h,float* upd,int L,const MP& p){
  gemm_rm(st,gw,h,64,p.in_w,1,p.in_b,w.xz,256,L,256,64,0);
  k_cconv<<<grd((long)L*128),256,0,st>>>(w.xz,p.conv_w,p.conv_b,w.xi,L);
  gemm_rm(st,gw,w.xi,128,p.xproj_w,1,nullptr,w.dbl,36,L,36,128,0);
  k_dt<<<grd((long)L*128),256,0,st>>>(w.dbl,p.dt_w,p.dt_b,w.dt,L);
  k_scan<<<64,32,0,st>>>(w.dt,w.dbl,w.xi,p.A_log,p.D,w.y,L);
  k_mulsiluz<<<grd((long)L*128),256,0,st>>>(w.y,w.xz,L);
  gemm_rm(st,gw,w.y,128,p.out_w,1,p.out_b,upd,64,L,64,128,0);
}
static void mamba_stage(hipStream_t st,GW gw,MWS& w,float* xcf,int L,const MLayer* Ls){
  long n=(long)L*64;
  k_cf2rm<<<grd(n),256,0,st>>>(w.xf,xcf,64,L);
  for(int li=0; li<2; ++li){
    const MLayer& Lp=Ls[li];
    k_lnorm<<<L,64,0,st>>>(w.xf,w.hb,Lp.ln_g,Lp.ln_b,L);
    run_mamba(st,gw,w,w.hb,w.upd,L,Lp.fwd);
    k_addrows<<<grd(n),256,0,st>>>(w.xf,w.upd,L,64,0);
    k_flip<<<grd(n),256,0,st>>>(w.hf,w.hb,L,64);
    run_mamba(st,gw,w,w.hf,w.upd,L,Lp.bwd);
    k_addrows<<<grd(n),256,0,st>>>(w.xf,w.upd,L,64,1);
  }
  k_rm2cf<<<grd(n),256,0,st>>>(xcf,w.xf,64,L);
}
static void gn8(hipStream_t st,float* x,int C,int Msp,const float* g,const float* b,float* stats,int act){
  k_gn_stats<<<8,256,0,st>>>(x,stats,C,Msp);
  k_gn_apply<<<grd((long)C*Msp),256,0,st>>>(x,stats,g,b,C,Msp,act);
}

extern "C" void kernel_launch(void* const* d_in,const int* in_sizes,int n_in,
                              void* d_out,int out_size,void* d_ws,size_t ws_size,
                              hipStream_t stream){
  (void)in_sizes; (void)n_in; (void)out_size;
  const int Dm=96, TT=2, HH=96, WWd=96;
  const int M=TT*HH*WWd, M48=TT*48*48, M24=TT*24*24, M12=TT*12*12;
  const int Lmax=M48;

  // ---- inputs (jax pytree flatten: dict keys sorted recursively) ----
  int ii=0;
  auto nx=[&](){ return (const float*)d_in[ii++]; };
  const float* aux=nx(); const float* mainp=nx();
  const float *ah_b1=nx(),*ah_b2=nx(),*ah_w1=nx(),*ah_w2=nx();
  const float *bt_b=nx(),*bt_w=nx();
  struct DnP{ const float *b,*bt,*g,*w; };
  DnP down1{nx(),nx(),nx(),nx()}, down2{nx(),nx(),nx(),nx()}, down3{nx(),nx(),nx(),nx()};
  struct FusP{ const float *bt,*g,*w; };
  FusP fus1{nx(),nx(),nx()}, fus2{nx(),nx(),nx()};
  const float *gate_b=nx(),*gate_w=nx();
  const float *gc_kb=nx(),*gc_kw=nx(),*gc_tb1=nx(),*gc_tb2=nx(),*gc_tw1=nx(),*gc_tw2=nx();
  auto rdMP=[&](){ MP p; p.A_log=nx(); p.D=nx(); p.conv_b=nx(); p.conv_w=nx(); p.dt_b=nx();
                   p.dt_w=nx(); p.in_b=nx(); p.in_w=nx(); p.out_b=nx(); p.out_w=nx(); p.xproj_w=nx(); return p; };
  auto rdL=[&](){ MLayer l; l.bwd=rdMP(); l.fwd=rdMP(); l.ln_b=nx(); l.ln_g=nx(); return l; };
  MLayer m1[2],m2[2],m3[2];
  m1[0]=rdL(); m1[1]=rdL(); m2[0]=rdL(); m2[1]=rdL(); m3[0]=rdL(); m3[1]=rdL();
  const float *mh_b1=nx(),*mh_b2=nx(),*mh_w1=nx(),*mh_w2=nx();
  const float *moe_eb1=nx(),*moe_eb2=nx(),*moe_ew1=nx(),*moe_ew2=nx(),*moe_gb=nx(),*moe_gw=nx();
  const float *ms3_dw=nx(),*ms3_pw=nx(),*ms5_dw=nx(),*ms5_pw=nx(),*ms_fuse=nx();
  struct ResP{ const float *s1dw,*s1pw,*s2dw,*s2pw,*tw; };
  ResP res[4];
  for(int i=0;i<4;++i) res[i]=ResP{nx(),nx(),nx(),nx(),nx()};
  struct SftP{ const float *bh,*bs,*wh,*ws; };
  SftP sft1{nx(),nx(),nx(),nx()}, sft2{nx(),nx(),nx(),nx()};
  const float *skip_b=nx(),*skip_g=nx(),*skip_w=nx();
  const float *t_b1=nx(),*t_b2=nx(),*t_w1=nx(),*t_w2=nx();
  const float *up1_b=nx(),*up1_w=nx();
  DnP up2{nx(),nx(),nx(),nx()}, up3{nx(),nx(),nx(),nx()};

  // ---- workspace layout ----
  char* wsb=(char*)d_ws; size_t off=0;
  auto alloc=[&](size_t bytes)->void*{ void* p=wsb+off; off=(off+bytes+255)&~(size_t)255; return p; };
  GW gw; gw.wb=(unsigned short*)alloc(2u<<20); gw.tap=(int4*)alloc(64u<<10);
  auto fal=[&](long nfl)->float*{ return (float*)alloc((size_t)nfl*4); };
  float *mn=fal(M);
  float *b_a=fal((long)Dm*M), *b_m=fal((long)Dm*M), *b_sh=fal((long)Dm*M), *b_x=fal((long)Dm*M);
  float *b_skip=fal((long)Dm*M), *b_x120=fal((long)Dm*M);
  float *t1=fal((long)Dm*M), *t2=fal((long)Dm*M), *t3=fal((long)Dm*M);
  float *cat=fal((long)2*Dm*M);
  float *moe_h=fal((long)2*Dm*M), *moe_oe=fal((long)Dm*M), *moe_acc=fal((long)Dm*M);
  float *rfin=fal(M);
  float *x60=fal((long)64*M48), *x60u=fal((long)64*M48), *x60f=fal((long)64*M48);
  float *x30=fal((long)64*M24), *x30u=fal((long)64*M24), *x30f=fal((long)64*M24);
  float *x15=fal((long)64*M12);
  float *catS=fal((long)128*M48);
  float *stats=fal(16), *pool=fal(96), *gate=fal(4), *attn=fal(M), *ctx=fal(96), *t2v=fal(96);
  MWS mw; mw.xf=fal((long)Lmax*64); mw.hb=fal((long)Lmax*64); mw.hf=fal((long)Lmax*64);
  mw.upd=fal((long)Lmax*64); mw.xz=fal((long)Lmax*256); mw.xi=fal((long)Lmax*128);
  mw.dbl=fal((long)Lmax*36); mw.dt=fal((long)Lmax*128); mw.y=fal((long)Lmax*128);
  if (off > ws_size) return;

  const long nDM=(long)Dm*M;
  // ---- mn ----
  k_scale<<<grd(M),256,0,stream>>>(mn,mainp,1.f/11.f,M);
  // ---- aux head ----
  gemm_conv(stream,gw,aux ,ah_w1,0,ah_b1,t1 ,9 ,TT,HH,WWd,3,3,3,1,1,1,1,1,1,HH,WWd,Dm,1,1);
  gemm_conv(stream,gw,t1  ,ah_w2,0,ah_b2,b_a,Dm,TT,HH,WWd,3,3,3,1,1,1,1,1,1,HH,WWd,Dm,0,1);
  // ---- multi-scale ----
  k_dw<<<grd(nDM),256,0,stream>>>(b_a,ms3_dw,t1,Dm,TT,HH,WWd,3,3,3,1,1,1,0);
  gemm_conv(stream,gw,t1,ms3_pw,0,nullptr,cat        ,Dm,TT,HH,WWd,1,1,1,1,1,1,0,0,0,HH,WWd,Dm,1,1);
  k_dw<<<grd(nDM),256,0,stream>>>(b_a,ms5_dw,t1,Dm,TT,HH,WWd,5,5,5,2,2,2,0);
  gemm_conv(stream,gw,t1,ms5_pw,0,nullptr,cat+nDM    ,Dm,TT,HH,WWd,1,1,1,1,1,1,0,0,0,HH,WWd,Dm,1,1);
  gemm_conv(stream,gw,cat,ms_fuse,0,nullptr,t1,2*Dm,TT,HH,WWd,1,1,1,1,1,1,0,0,0,HH,WWd,Dm,0,1);
  k_add<<<grd(nDM),256,0,stream>>>(b_a,t1,(int)nDM);
  // ---- main head ----
  gemm_conv(stream,gw,mn,mh_w1,0,mh_b1,t1 ,1 ,TT,HH,WWd,3,3,3,1,1,1,1,1,1,HH,WWd,Dm,1,1);
  gemm_conv(stream,gw,t1,mh_w2,0,mh_b2,b_m,Dm,TT,HH,WWd,3,3,3,1,1,1,1,1,1,HH,WWd,Dm,0,1);
  k_copy<<<grd(nDM),256,0,stream>>>(b_sh,b_m,(int)nDM);
  // ---- sft1 + gate fuse ----
  gemm_conv(stream,gw,b_a,sft1.ws,0,sft1.bs,t1,Dm,TT,HH,WWd,1,1,1,1,1,1,0,0,0,HH,WWd,Dm,0,1);
  gemm_conv(stream,gw,b_a,sft1.wh,0,sft1.bh,t2,Dm,TT,HH,WWd,1,1,1,1,1,1,0,0,0,HH,WWd,Dm,0,1);
  k_sft<<<grd(nDM),256,0,stream>>>(b_x,b_m,t1,t2,(int)nDM);
  k_copy<<<grd(nDM),256,0,stream>>>(cat,b_x,(int)nDM);
  k_copy<<<grd(nDM),256,0,stream>>>(cat+nDM,b_a,(int)nDM);
  gemm_conv(stream,gw,cat,gate_w,0,gate_b,t1,2*Dm,TT,HH,WWd,1,1,1,1,1,1,0,0,0,HH,WWd,Dm,2,1);
  k_blend<<<grd(nDM),256,0,stream>>>(b_x,t1,b_a,(int)nDM);
  // ---- residual blocks ----
  for(int rb=0;rb<4;++rb){
    k_dw<<<grd(nDM),256,0,stream>>>(b_x,res[rb].tw,t1,Dm,TT,HH,WWd,3,1,1,1,0,0,1);
    k_dw<<<grd(nDM),256,0,stream>>>(t1,res[rb].s1dw,t2,Dm,TT,HH,WWd,3,3,3,1,1,1,0);
    gemm_conv(stream,gw,t2,res[rb].s1pw,0,nullptr,t3,Dm,TT,HH,WWd,1,1,1,1,1,1,0,0,0,HH,WWd,Dm,1,1);
    k_dw<<<grd(nDM),256,0,stream>>>(t3,res[rb].s2dw,t1,Dm,TT,HH,WWd,3,3,3,1,1,1,0);
    gemm_conv(stream,gw,t1,res[rb].s2pw,0,nullptr,t2,Dm,TT,HH,WWd,1,1,1,1,1,1,0,0,0,HH,WWd,Dm,1,1);
    k_add<<<grd(nDM),256,0,stream>>>(b_x,t2,(int)nDM);
  }
  // ---- bottleneck + skip ----
  gemm_conv(stream,gw,b_x,bt_w,0,bt_b,t1,Dm,TT,HH,WWd,3,3,3,1,1,1,1,1,1,HH,WWd,Dm,0,1);
  k_add2<<<grd(nDM),256,0,stream>>>(b_x,t1,b_sh,(int)nDM);
  k_copy<<<grd(nDM),256,0,stream>>>(b_skip,b_x,(int)nDM);
  // ---- encoder ----
  gemm_conv(stream,gw,b_x,down1.w,0,down1.b,x60,Dm,TT,HH,WWd,1,3,3,1,2,2,0,1,1,48,48,64,0,1);
  gn8(stream,x60,64,M48,down1.g,down1.bt,stats,1);
  mamba_stage(stream,gw,mw,x60,M48,m1);
  gemm_conv(stream,gw,x60,down2.w,0,down2.b,x30,64,TT,48,48,1,3,3,1,2,2,0,1,1,24,24,64,0,1);
  gn8(stream,x30,64,M24,down2.g,down2.bt,stats,1);
  mamba_stage(stream,gw,mw,x30,M24,m2);
  gemm_conv(stream,gw,x30,down3.w,0,down3.b,x15,64,TT,24,24,1,3,3,1,2,2,0,1,1,12,12,64,0,1);
  gn8(stream,x15,64,M12,down3.g,down3.bt,stats,1);
  mamba_stage(stream,gw,mw,x15,M12,m3);
  // ---- decoder ----
  gemm_conv(stream,gw,x15,up3.w,2,up3.b,x30u,64,TT,12,12,1,4,4,1,2,2,0,1,1,24,24,64,0,2);
  gn8(stream,x30u,64,M24,up3.g,up3.bt,stats,1);
  k_copy<<<grd((long)64*M24),256,0,stream>>>(catS,x30u,64*M24);
  k_copy<<<grd((long)64*M24),256,0,stream>>>(catS+(long)64*M24,x30,64*M24);
  gemm_conv(stream,gw,catS,fus2.w,0,nullptr,x30f,128,TT,24,24,1,1,1,1,1,1,0,0,0,24,24,64,0,1);
  gn8(stream,x30f,64,M24,fus2.g,fus2.bt,stats,1);
  gemm_conv(stream,gw,x30f,up2.w,2,up2.b,x60u,64,TT,24,24,1,4,4,1,2,2,0,1,1,48,48,64,0,2);
  gn8(stream,x60u,64,M48,up2.g,up2.bt,stats,1);
  k_copy<<<grd((long)64*M48),256,0,stream>>>(catS,x60u,64*M48);
  k_copy<<<grd((long)64*M48),256,0,stream>>>(catS+(long)64*M48,x60,64*M48);
  gemm_conv(stream,gw,catS,fus1.w,0,nullptr,x60f,128,TT,48,48,1,1,1,1,1,1,0,0,0,48,48,64,0,1);
  gn8(stream,x60f,64,M48,fus1.g,fus1.bt,stats,1);
  gemm_conv(stream,gw,x60f,up1_w,2,up1_b,b_x120,64,TT,48,48,1,4,4,1,2,2,0,1,1,HH,WWd,Dm,0,2);
  // ---- skip fuse ----
  k_copy<<<grd(nDM),256,0,stream>>>(cat,b_skip,(int)nDM);
  k_copy<<<grd(nDM),256,0,stream>>>(cat+nDM,b_x120,(int)nDM);
  gemm_conv(stream,gw,cat,skip_w,0,nullptr,b_x,2*Dm,TT,HH,WWd,1,1,1,1,1,1,0,0,0,HH,WWd,Dm,0,1);
  gn8(stream,b_x,Dm,M,skip_g,skip_b,stats,1);
  // ---- MoE ----
  k_chan_mean<<<Dm,256,0,stream>>>(b_x,pool,M);
  k_moe_gate<<<1,32,0,stream>>>(pool,moe_gw,moe_gb,gate);
  k_fill<<<grd(nDM),256,0,stream>>>(moe_acc,0.f,(int)nDM);
  for(int e=0;e<4;++e){
    gemm_conv(stream,gw,b_x,moe_ew1+(size_t)e*192*96,0,moe_eb1+e*192,moe_h,Dm,TT,HH,WWd,1,1,1,1,1,1,0,0,0,HH,WWd,192,1,1);
    gemm_conv(stream,gw,moe_h,moe_ew2+(size_t)e*96*192,0,moe_eb2+e*96,moe_oe,192,TT,HH,WWd,1,1,1,1,1,1,0,0,0,HH,WWd,Dm,0,1);
    k_axpyg<<<grd(nDM),256,0,stream>>>(moe_acc,moe_oe,gate,e,(int)nDM);
  }
  k_add<<<grd(nDM),256,0,stream>>>(b_x,moe_acc,(int)nDM);
  // ---- global context ----
  k_kproj<<<grd(M),256,0,stream>>>(b_x,gc_kw,gc_kb,attn,Dm,M);
  k_softmax1<<<1,256,0,stream>>>(attn,M);
  k_chan_dot<<<Dm,256,0,stream>>>(b_x,attn,ctx,M);
  k_gc_mlp<<<1,128,0,stream>>>(ctx,gc_tw1,gc_tb1,gc_tw2,gc_tb2,t2v);
  k_addvec_cf<<<grd(nDM),256,0,stream>>>(b_x,t2v,M,(int)nDM);
  // ---- sft2 + tail ----
  gemm_conv(stream,gw,b_a,sft2.ws,0,sft2.bs,t1,Dm,TT,HH,WWd,1,1,1,1,1,1,0,0,0,HH,WWd,Dm,0,1);
  gemm_conv(stream,gw,b_a,sft2.wh,0,sft2.bh,t2,Dm,TT,HH,WWd,1,1,1,1,1,1,0,0,0,HH,WWd,Dm,0,1);
  k_sft<<<grd(nDM),256,0,stream>>>(b_x,b_x,t1,t2,(int)nDM);
  gemm_conv(stream,gw,b_x,t_w1,0,t_b1,t1,Dm,TT,HH,WWd,3,3,3,1,1,1,1,1,1,HH,WWd,Dm,1,1);
  gemm_conv(stream,gw,t1,t_w2,0,t_b2,rfin,Dm,TT,HH,WWd,1,1,1,1,1,1,0,0,0,HH,WWd,1,0,1);
  k_final<<<grd(M),256,0,stream>>>((float*)d_out,mainp,rfin,aux,M);
}
